// ScaledDotAttention_31327491457657
// MI455X (gfx1250) — matrix operand order repaired and checked by reading, not compiled, not run on hardware
//
#include <hip/hip_runtime.h>

typedef __attribute__((ext_vector_type(16))) _Float16 v16h;
typedef __attribute__((ext_vector_type(8)))  _Float16 v8h;
typedef __attribute__((ext_vector_type(4)))  _Float16 h4;
typedef __attribute__((ext_vector_type(8)))  float    v8f;
typedef __attribute__((ext_vector_type(4)))  unsigned int u32x4;
typedef __attribute__((ext_vector_type(8)))  int      i32x8;
typedef __attribute__((ext_vector_type(4)))  int      i32x4;

#define B_  2
#define H_  16
#define S_  2048
#define D_  64
#define BQ  128   // q rows per block
#define BK  64    // k rows per tile iteration (4 score tiles)
#define NW  8     // waves per block (wave32)

// ---------------- fragment helpers ----------------
__device__ __forceinline__ v16h ld16(const _Float16* p) {
  v8h lo = *(const v8h*)p;
  v8h hi = *(const v8h*)(p + 8);
  v16h r;
#pragma unroll
  for (int i = 0; i < 8; ++i) { r[i] = lo[i]; r[i + 8] = hi[i]; }
  return r;
}
__device__ __forceinline__ v16h ld8x2(const _Float16* p0, const _Float16* p1) {
  v8h lo = *(const v8h*)p0;
  v8h hi = *(const v8h*)p1;
  v16h r;
#pragma unroll
  for (int i = 0; i < 8; ++i) { r[i] = lo[i]; r[i + 8] = hi[i]; }
  return r;
}
__device__ __forceinline__ float rowmax16(float v) {
#pragma unroll
  for (int m = 8; m >= 1; m >>= 1) v = fmaxf(v, __shfl_xor(v, m, 32));
  return v;
}
__device__ __forceinline__ float rowsum16(float v) {
#pragma unroll
  for (int m = 8; m >= 1; m >>= 1) v += __shfl_xor(v, m, 32);
  return v;
}

// ---------------- TDM issue (2-D tile, f16 elements) ----------------
#if defined(__HIP_DEVICE_COMPILE__) && __has_builtin(__builtin_amdgcn_tensor_load_to_lds)
#define HAVE_TDM 1
__device__ __forceinline__ void tdm_load_2d(unsigned int lds_addr, const void* gptr,
                                            unsigned int td0, unsigned int td1,
                                            unsigned int tile0, unsigned int tile1,
                                            unsigned long long stride0) {
  const unsigned long long ga = (unsigned long long)(uintptr_t)gptr;
  // D# group 0: count=1 | lds_addr | global_addr[56:0] | type=2
  u32x4 g0 = { 1u,
               lds_addr,
               (unsigned int)ga,
               (unsigned int)((ga >> 32) & 0x01FFFFFFu) | (2u << 30) };
  // D# group 1: data_size=2B; tensor_dim0/1, tile_dim0/1, tensor_dim0_stride
  i32x8 g1 = { (int)(1u << 16),                                     // data_size=1 (2 bytes)
               (int)((td0 & 0xFFFFu) << 16),                        // tensor_dim0[15:0]
               (int)(((td0 >> 16) & 0xFFFFu) | ((td1 & 0xFFFFu) << 16)),
               (int)(((td1 >> 16) & 0xFFFFu) | ((tile0 & 0xFFFFu) << 16)),
               (int)(tile1 & 0xFFFFu),                              // tile_dim1, tile_dim2=0
               (int)(unsigned int)(stride0 & 0xFFFFFFFFu),
               (int)(unsigned int)((stride0 >> 32) & 0xFFFFu),      // dim1_stride unused (2-D)
               0 };
  i32x4 z4 = { 0, 0, 0, 0 };
#if defined(__clang_major__) && (__clang_major__ >= 23)
  i32x8 z8 = { 0, 0, 0, 0, 0, 0, 0, 0 };
  __builtin_amdgcn_tensor_load_to_lds(g0, g1, z4, z4, z8, 0);
#else
  __builtin_amdgcn_tensor_load_to_lds(g0, g1, z4, z4, 0);
#endif
}
#else
#define HAVE_TDM 0
#endif

// ---------------- K/V fp32 -> f16 convert (V transposed via LDS tile) ----------------
__global__ __launch_bounds__(256, 2)
void cvt_kv(const float* __restrict__ Kg, const float* __restrict__ Vg,
            _Float16* __restrict__ Kh, _Float16* __restrict__ Vth) {
  __shared__ _Float16 T[64][68];                 // padded to dodge bank conflicts
  const int tid = threadIdx.x;
  const int bh  = blockIdx.y;
  const int sb  = blockIdx.x * 64;
#pragma unroll
  for (int i = 0; i < 4; ++i) {
    const int t    = tid + i * 256;              // 0..1023 -> 64 rows x 16 float4
    const int srow = t >> 4;
    const int d4   = (t & 15) << 2;
    const size_t g = ((size_t)bh * S_ + (size_t)(sb + srow)) * D_ + d4;
    const float4 k4 = *(const float4*)(Kg + g);
    const float4 v4 = *(const float4*)(Vg + g);
    h4 kh = { (_Float16)k4.x, (_Float16)k4.y, (_Float16)k4.z, (_Float16)k4.w };
    *(h4*)(Kh + g) = kh;                         // K: same layout, coalesced
    T[d4 + 0][srow] = (_Float16)v4.x;            // V: transpose through LDS
    T[d4 + 1][srow] = (_Float16)v4.y;
    T[d4 + 2][srow] = (_Float16)v4.z;
    T[d4 + 3][srow] = (_Float16)v4.w;
  }
  __syncthreads();
#pragma unroll
  for (int i = 0; i < 4; ++i) {
    const int t  = tid + i * 256;                // 0..1023 -> 64 d-rows x 16 s4
    const int d  = t >> 4;
    const int s4 = (t & 15) << 2;
    h4 o = { T[d][s4], T[d][s4 + 1], T[d][s4 + 2], T[d][s4 + 3] };
    *(h4*)(Vth + ((size_t)bh * D_ + d) * S_ + (size_t)(sb + s4)) = o;
  }
}

// ---------------- flash attention, f16 K/V staged by the Tensor Data Mover ----------------
__global__ __launch_bounds__(256, 1)
void fa_tdm(const float* __restrict__ Qg, const _Float16* __restrict__ Kh,
            const _Float16* __restrict__ Vth, float* __restrict__ Og) {
  __shared__ __attribute__((aligned(16))) _Float16 Kbuf[2][BK][D_];   // (key, d)
  __shared__ __attribute__((aligned(16))) _Float16 Vtbuf[2][D_][BK];  // (d, key)
  __shared__ __attribute__((aligned(16))) _Float16 Plds[NW][16][BK];

  const int tid   = threadIdx.x;
  const int lane  = tid & 31;
  const int wave  = tid >> 5;
  const int col   = lane & 15;
  const int hsel  = (lane < 16) ? 0 : 1;
  const int aoff  = hsel * 8;
  const int boff  = hsel * 16;
  const int rbase = hsel * 8;
  const int bh    = blockIdx.y;
  const int qb    = blockIdx.x * BQ;
  const int qbase = qb + wave * 16;
  const int qhi   = qbase + 15;
  const float NEG_INF = -__builtin_inff();

  // Q fragment (A layout), fp32 -> f16
  const float* qptr = Qg + ((size_t)bh * S_ + (qbase + col)) * D_;
  v16h qa[2];
#pragma unroll
  for (int c = 0; c < 2; ++c) {
#pragma unroll
    for (int e = 0; e < 8; ++e) {
      qa[c][e]     = (_Float16)qptr[c * 32 + aoff + e];
      qa[c][e + 8] = (_Float16)qptr[c * 32 + 16 + aoff + e];
    }
  }

  v8f o[4] = {};
  float mrow[8], lpart[8];
#pragma unroll
  for (int r = 0; r < 8; ++r) { mrow[r] = NEG_INF; lpart[r] = 0.f; }

  const int nkb = (qb + BQ) / BK;

#if HAVE_TDM
  // prologue: DMA tile 0 into buffer 0 (wave 0 only; TENSORcnt tracks it)
  if (wave == 0) {
    tdm_load_2d((unsigned int)(uintptr_t)&Kbuf[0][0][0],
                Kh + ((size_t)bh * S_) * D_, D_, S_, D_, BK, D_);
    tdm_load_2d((unsigned int)(uintptr_t)&Vtbuf[0][0][0],
                Vth + (size_t)bh * D_ * S_, S_, D_, BK, D_, S_);
  }
#endif

  for (int kb = 0; kb < nkb; ++kb) {
    const int buf = kb & 1;
#if HAVE_TDM
    if (wave == 0) __builtin_amdgcn_s_wait_tensorcnt((short)0);  // tile kb landed
    __syncthreads();                                             // publish to all waves
    if (wave == 0 && kb + 1 < nkb) {                             // DMA next tile into
      const int nb = buf ^ 1;                                    // the retired buffer
      tdm_load_2d((unsigned int)(uintptr_t)&Kbuf[nb][0][0],
                  Kh + ((size_t)bh * S_ + (size_t)((kb + 1) * BK)) * D_,
                  D_, S_, D_, BK, D_);
      tdm_load_2d((unsigned int)(uintptr_t)&Vtbuf[nb][0][0],
                  Vth + (size_t)bh * D_ * S_ + (size_t)((kb + 1) * BK),
                  S_, D_, BK, D_, S_);
    }
#else
    // fallback: cooperative f16 copy (no TDM builtin on this toolchain)
    __syncthreads();
#pragma unroll
    for (int i = 0; i < 2; ++i) {
      const int t   = tid + i * 256;     // 0..511 -> 64 rows x 8 segs of 8 halfs
      const int row = t >> 3;
      const int seg = (t & 7) * 8;
      *(v8h*)&Kbuf[buf][row][seg] =
          *(const v8h*)(Kh + ((size_t)bh * S_ + (size_t)(kb * BK + row)) * D_ + seg);
      *(v8h*)&Vtbuf[buf][row][seg] =
          *(const v8h*)(Vth + ((size_t)bh * D_ + row) * S_ + (size_t)(kb * BK + seg));
    }
    __syncthreads();
#endif

    const int k0 = kb * BK;
    if (k0 <= qhi) {
      v8f s[4] = {};
#pragma unroll
      for (int c = 0; c < 2; ++c) {
#pragma unroll
        for (int t = 0; t < 4; ++t) {
          v16h b = ld8x2(&Kbuf[buf][t * 16 + col][c * 32 + aoff], &Kbuf[buf][t * 16 + col][c * 32 + 16 + aoff]);
          s[t] = __builtin_amdgcn_wmma_f32_16x16x32_f16(false, qa[c], false, b,
                                                        (short)0, s[t], false, false);
        }
      }

      if (k0 + BK - 1 > qbase) {     // diagonal tile only
#pragma unroll
        for (int t = 0; t < 4; ++t) {
          const int kidx = k0 + t * 16 + col;
#pragma unroll
          for (int r = 0; r < 8; ++r)
            s[t][r] = (kidx <= qbase + rbase + r) ? s[t][r] : NEG_INF;
        }
      }

#pragma unroll
      for (int r = 0; r < 8; ++r) {
        const float tm = rowmax16(fmaxf(fmaxf(s[0][r], s[1][r]),
                                        fmaxf(s[2][r], s[3][r])));
        const float mn = fmaxf(mrow[r], tm);
        const float sc = __expf(mrow[r] - mn);
        float p0 = __expf(s[0][r] - mn);
        float p1 = __expf(s[1][r] - mn);
        float p2 = __expf(s[2][r] - mn);
        float p3 = __expf(s[3][r] - mn);
        lpart[r] = lpart[r] * sc + ((p0 + p1) + (p2 + p3));
        mrow[r] = mn;
        o[0][r] *= sc; o[1][r] *= sc; o[2][r] *= sc; o[3][r] *= sc;
        _Float16* pr = &Plds[wave][rbase + r][0];
        pr[col]      = (_Float16)p0;
        pr[16 + col] = (_Float16)p1;
        pr[32 + col] = (_Float16)p2;
        pr[48 + col] = (_Float16)p3;
      }

      const _Float16* prow = &Plds[wave][col][0];
#pragma unroll
      for (int c = 0; c < 2; ++c) {
        v16h pa = ld8x2(prow + c * 32 + aoff, prow + c * 32 + 16 + aoff);
#pragma unroll
        for (int j = 0; j < 4; ++j) {
          v16h vb = ld8x2(&Vtbuf[buf][j * 16 + col][c * 32 + aoff], &Vtbuf[buf][j * 16 + col][c * 32 + 16 + aoff]);
          o[j] = __builtin_amdgcn_wmma_f32_16x16x32_f16(false, pa, false, vb,
                                                        (short)0, o[j], false, false);
        }
      }
    }
  }

#pragma unroll
  for (int r = 0; r < 8; ++r) {
    const float rinv = 1.0f / rowsum16(lpart[r]);
    const size_t orow = ((size_t)bh * S_ + (size_t)(qbase + rbase + r)) * D_;
#pragma unroll
    for (int j = 0; j < 4; ++j)
      Og[orow + j * 16 + col] = o[j][r] * rinv;
  }
}

// ---------------- proven fp32-input path (fallback when ws is too small) ----------------
__global__ __launch_bounds__(256, 1)
void fa_fwd_causal(const float* __restrict__ Qg, const float* __restrict__ Kg,
                   const float* __restrict__ Vg, float* __restrict__ Og) {
  __shared__ __attribute__((aligned(16))) _Float16 Klds[BK][D_];
  __shared__ __attribute__((aligned(16))) _Float16 Vtlds[D_][BK];
  __shared__ __attribute__((aligned(16))) _Float16 Plds[NW][16][BK];

  const int tid = threadIdx.x, lane = tid & 31, wave = tid >> 5;
  const int col = lane & 15, hsel = (lane < 16) ? 0 : 1;
  const int aoff = hsel * 8, boff = hsel * 16, rbase = hsel * 8;
  const int bh = blockIdx.y, qb = blockIdx.x * BQ;
  const int qbase = qb + wave * 16, qhi = qbase + 15;
  const float NEG_INF = -__builtin_inff();

  const float* qptr = Qg + ((size_t)bh * S_ + (qbase + col)) * D_;
  v16h qa[2];
#pragma unroll
  for (int c = 0; c < 2; ++c)
#pragma unroll
    for (int e = 0; e < 8; ++e) {
      qa[c][e]     = (_Float16)qptr[c * 32 + aoff + e];
      qa[c][e + 8] = (_Float16)qptr[c * 32 + 16 + aoff + e];
    }

  v8f o[4] = {};
  float mrow[8], lpart[8];
#pragma unroll
  for (int r = 0; r < 8; ++r) { mrow[r] = NEG_INF; lpart[r] = 0.f; }

  const int nkb = (qb + BQ) / BK;
  for (int kb = 0; kb < nkb; ++kb) {
#pragma unroll
    for (int i = 0; i < 4; ++i) {
      const int f = tid + i * 256, row = f >> 4, c4 = (f & 15) << 2;
      const size_t gbase = ((size_t)bh * S_ + (size_t)(kb * BK + row)) * D_ + c4;
      const float4 kv4 = *(const float4*)(Kg + gbase);
      const float4 vv4 = *(const float4*)(Vg + gbase);
      Klds[row][c4 + 0] = (_Float16)kv4.x; Klds[row][c4 + 1] = (_Float16)kv4.y;
      Klds[row][c4 + 2] = (_Float16)kv4.z; Klds[row][c4 + 3] = (_Float16)kv4.w;
      Vtlds[c4 + 0][row] = (_Float16)vv4.x; Vtlds[c4 + 1][row] = (_Float16)vv4.y;
      Vtlds[c4 + 2][row] = (_Float16)vv4.z; Vtlds[c4 + 3][row] = (_Float16)vv4.w;
    }
    __syncthreads();
    const int k0 = kb * BK;
    if (k0 <= qhi) {
      v8f s[4] = {};
#pragma unroll
      for (int c = 0; c < 2; ++c)
#pragma unroll
        for (int t = 0; t < 4; ++t) {
          v16h b = ld8x2(&Klds[t * 16 + col][c * 32 + aoff], &Klds[t * 16 + col][c * 32 + 16 + aoff]);
          s[t] = __builtin_amdgcn_wmma_f32_16x16x32_f16(false, qa[c], false, b,
                                                        (short)0, s[t], false, false);
        }
      if (k0 + BK - 1 > qbase) {
#pragma unroll
        for (int t = 0; t < 4; ++t) {
          const int kidx = k0 + t * 16 + col;
#pragma unroll
          for (int r = 0; r < 8; ++r)
            s[t][r] = (kidx <= qbase + rbase + r) ? s[t][r] : NEG_INF;
        }
      }
#pragma unroll
      for (int r = 0; r < 8; ++r) {
        const float tm = rowmax16(fmaxf(fmaxf(s[0][r], s[1][r]),
                                        fmaxf(s[2][r], s[3][r])));
        const float mn = fmaxf(mrow[r], tm);
        const float sc = __expf(mrow[r] - mn);
        float p0 = __expf(s[0][r] - mn), p1 = __expf(s[1][r] - mn);
        float p2 = __expf(s[2][r] - mn), p3 = __expf(s[3][r] - mn);
        lpart[r] = lpart[r] * sc + ((p0 + p1) + (p2 + p3));
        mrow[r] = mn;
        o[0][r] *= sc; o[1][r] *= sc; o[2][r] *= sc; o[3][r] *= sc;
        _Float16* pr = &Plds[wave][rbase + r][0];
        pr[col] = (_Float16)p0; pr[16 + col] = (_Float16)p1;
        pr[32 + col] = (_Float16)p2; pr[48 + col] = (_Float16)p3;
      }
      const _Float16* prow = &Plds[wave][col][0];
#pragma unroll
      for (int c = 0; c < 2; ++c) {
        v16h pa = ld8x2(prow + c * 32 + aoff, prow + c * 32 + 16 + aoff);
#pragma unroll
        for (int j = 0; j < 4; ++j) {
          v16h vb = ld8x2(&Vtlds[j * 16 + col][c * 32 + aoff], &Vtlds[j * 16 + col][c * 32 + 16 + aoff]);
          o[j] = __builtin_amdgcn_wmma_f32_16x16x32_f16(false, pa, false, vb,
                                                        (short)0, o[j], false, false);
        }
      }
    }
    __syncthreads();
  }
#pragma unroll
  for (int r = 0; r < 8; ++r) {
    const float rinv = 1.0f / rowsum16(lpart[r]);
    const size_t orow = ((size_t)bh * S_ + (size_t)(qbase + rbase + r)) * D_;
#pragma unroll
    for (int j = 0; j < 4; ++j)
      Og[orow + j * 16 + col] = o[j][r] * rinv;
  }
}

extern "C" void kernel_launch(void* const* d_in, const int* in_sizes, int n_in,
                              void* d_out, int out_size, void* d_ws, size_t ws_size,
                              hipStream_t stream) {
  (void)in_sizes; (void)n_in; (void)out_size;
  const float* q = (const float*)d_in[0];
  const float* k = (const float*)d_in[1];
  const float* v = (const float*)d_in[2];
  float* out = (float*)d_out;
  dim3 grid(S_ / BQ, B_ * H_, 1);
  dim3 block(256, 1, 1);

  const size_t half_elems = (size_t)B_ * H_ * S_ * D_;
  const size_t need = half_elems * sizeof(_Float16) * 2;   // f16 K + transposed f16 V
  if (ws_size >= need) {
    _Float16* Kh  = (_Float16*)d_ws;
    _Float16* Vth = Kh + half_elems;
    cvt_kv<<<dim3(S_ / 64, B_ * H_, 1), block, 0, stream>>>(k, v, Kh, Vth);
    fa_tdm<<<grid, block, 0, stream>>>(q, Kh, Vth, out);
  } else {
    fa_fwd_causal<<<grid, block, 0, stream>>>(q, k, v, out);
  }
}
